// Relate_19361712570516
// MI455X (gfx1250) — compile-verified, matrix operand order checked by reading, not run on hardware
//
#include <hip/hip_runtime.h>
#include <hip/hip_bf16.h>
#include <math.h>

// ---------------------------------------------------------------------------
// Types & WMMA helpers (CDNA5 gfx1250, wave32, v_wmma_f32_16x16x32_bf16)
// ---------------------------------------------------------------------------
typedef __attribute__((ext_vector_type(16))) __bf16 v16bf;
typedef __attribute__((ext_vector_type(8)))  float  v8f;

union Frag { v16bf bf; uint4 u[2]; };

__device__ __forceinline__ unsigned short f2bf(float f) {
  unsigned int u = __float_as_uint(f);
  u += 0x7FFFu + ((u >> 16) & 1u);          // round-to-nearest-even
  return (unsigned short)(u >> 16);
}

__device__ __forceinline__ v8f wmma_bf16(v16bf a, v16bf b, v8f c) {
  return __builtin_amdgcn_wmma_f32_16x16x32_bf16(
      /*neg_a=*/false, a, /*neg_b=*/false, b,
      /*c_mod=*/(short)0, c, /*reuse_a=*/false, /*reuse_b=*/false);
}

// A-fragment (or B-fragment whose K-dim is contiguous in memory) from a
// row-major bf16 buffer. CDNA5 16-bit A layout: lane L (row = L&15, half
// h = L>>4) holds K = k0 + h*8 .. +7 in elems 0..7 and K = k0+16+h*8 .. +7
// in elems 8..15 -> two aligned 16-byte loads per lane.
__device__ __forceinline__ v16bf load_frag_rm(const unsigned short* base,
                                              int stride, int k0, int lane) {
  int r = lane & 15, hh = lane >> 4;
  const unsigned short* p = base + (size_t)r * stride + k0 + hh * 8;
  Frag f;
  f.u[0] = *(const uint4*)(p);
  f.u[1] = *(const uint4*)(p + 16);
  return f.bf;
}

// Weight B-fragment from a pre-packed buffer: layout [nt][kt][lane][16 bf16],
// so each lane reads 32 contiguous bytes (wave reads 1KB contiguous).
__device__ __forceinline__ v16bf load_frag_packed(const unsigned short* pack,
                                                  int KC, int nt, int kt,
                                                  int lane) {
  const unsigned short* p = pack + (((size_t)nt * KC + kt) * 32 + lane) * 16;
  Frag f;
  f.u[0] = ((const uint4*)p)[0];
  f.u[1] = ((const uint4*)p)[1];
  return f.bf;
}

// ---------------------------------------------------------------------------
// Prep kernels
// ---------------------------------------------------------------------------
__global__ void k_cvt_bf16(const float* __restrict__ src,
                           unsigned short* __restrict__ dst, int n) {
  int t = blockIdx.x * blockDim.x + threadIdx.x;
  if (t < n) dst[t] = f2bf(src[t]);
}

// Pack f32 weight W[K x N] (row stride ld) into WMMA B-fragment layout.
__global__ void k_pack_w(const float* __restrict__ W, int ld, int K, int N,
                         unsigned short* __restrict__ out) {
  int t = blockIdx.x * blockDim.x + threadIdx.x;
  if (t >= N * K) return;
  int e     = t & 15;
  int lane  = (t >> 4) & 31;
  int blkid = t >> 9;
  int KC = K >> 5;
  int kt = blkid % KC;
  int nt = blkid / KC;
  int n  = nt * 16 + (lane & 15);
  int hh = lane >> 4;
  int k  = kt * 32 + ((e < 8) ? 0 : 16) + hh * 8 + (e & 7);
  out[t] = f2bf(W[(size_t)k * ld + n]);
}

// q = (guides@Wq + bq) * D^-0.5, stored bf16 layout [g, h, d] (= [g, vd]).
__global__ void k_qproj(const float* __restrict__ guides,
                        const float* __restrict__ Wq,
                        const float* __restrict__ bq,
                        unsigned short* __restrict__ q_bf) {
  int t = blockIdx.x * blockDim.x + threadIdx.x;
  if (t >= 16 * 1024) return;
  int g = t >> 10, vd = t & 1023;
  float acc = bq[vd];
  for (int dd = 0; dd < 128; ++dd)
    acc += guides[g * 128 + dd] * Wq[dd * 1024 + vd];
  q_bf[t] = f2bf(acc * 0.08838834764831845f);  // fold 1/sqrt(128) into q
}

__global__ void k_zero(float* p, int n) {
  int t = blockIdx.x * blockDim.x + threadIdx.x;
  if (t < n) p[t] = 0.f;
}

// ---------------------------------------------------------------------------
// K/V projection GEMM: [8192,128] @ packed[128,1024] (+bias), bf16 WMMA.
// mode 0: out[b,h,s,d] (K tensor)   mode 1: out[b,h,d,s] (V transposed)
// Block = 16 rows x 1024 cols, 8 waves x 8 col-tiles each.
// ---------------------------------------------------------------------------
__global__ void k_proj(const unsigned short* __restrict__ x_bf,
                       const unsigned short* __restrict__ Wpack,
                       const float* __restrict__ bias,
                       unsigned short* __restrict__ outp, int mode) {
  const int lane = threadIdx.x & 31;
  const int wv   = threadIdx.x >> 5;
  const int m0   = blockIdx.x * 16;
  v8f acc[8];
  #pragma unroll
  for (int t = 0; t < 8; ++t)
    for (int z = 0; z < 8; ++z) acc[t][z] = 0.f;

  const unsigned short* abase = x_bf + (size_t)m0 * 128;
  for (int kt = 0; kt < 4; ++kt) {
    // batch all loads of this K-chunk first, then issue the 8 WMMAs,
    // so the backend can use partial s_wait_loadcnt instead of wait-0.
    v16bf a = load_frag_rm(abase, 128, kt * 32, lane);
    v16bf b[8];
    #pragma unroll
    for (int t = 0; t < 8; ++t)
      b[t] = load_frag_packed(Wpack, 4, wv * 8 + t, kt, lane);
    #pragma unroll
    for (int t = 0; t < 8; ++t)
      acc[t] = wmma_bf16(a, b[t], acc[t]);
  }
  const int laneh = lane >> 4, lanel = lane & 15;
  #pragma unroll
  for (int t = 0; t < 8; ++t) {
    int vd = (wv * 8 + t) * 16 + lanel;
    int head = vd >> 7, dd = vd & 127;
    float bv = bias[vd];
    #pragma unroll
    for (int r = 0; r < 8; ++r) {
      int m = m0 + r + 8 * laneh;      // global (b,s) row
      int bb = m >> 10, s = m & 1023;
      unsigned short val = f2bf(acc[t][r] + bv);
      if (mode == 0)
        outp[(((size_t)bb * 8 + head) * 1024 + s) * 128 + dd] = val;
      else
        outp[(((size_t)bb * 8 + head) * 128 + dd) * 1024 + s] = val;
    }
  }
}

// ---------------------------------------------------------------------------
// Fused attention per (b,h): scores=q@K^T (WMMA) -> softmax (LDS) ->
// o=attn@V (WMMA) -> head_wise bf16 [b, n=g*H+h, d].
// ---------------------------------------------------------------------------
__global__ void k_attn(const unsigned short* __restrict__ q_bf,
                       const unsigned short* __restrict__ k_bf,
                       const unsigned short* __restrict__ v_t,
                       unsigned short* __restrict__ hw_bf) {
  extern __shared__ char smem[];
  float*          sc      = (float*)smem;                               // 16 x 1032 f32
  unsigned short* attn    = (unsigned short*)(smem + 16 * 1032 * 4);    // 16 x 1040 bf16
  float*          red     = (float*)(smem + 16 * 1032 * 4 + 16 * 1040 * 2); // 16x16
  float*          rowstat = red + 256;                                  // 16 max + 16 sum

  const int bid = blockIdx.x;
  const int bb = bid >> 3, hh = bid & 7;
  const int lane = threadIdx.x & 31, wv = threadIdx.x >> 5;
  const int laneh = lane >> 4, lanel = lane & 15;

  // ---- scores: M=16(g), N=1024(s), K=128(d); 8 waves x 8 s-tiles
  {
    v8f acc[8];
    #pragma unroll
    for (int t = 0; t < 8; ++t)
      for (int z = 0; z < 8; ++z) acc[t][z] = 0.f;
    const unsigned short* qbase = q_bf + hh * 128;  // rows g, stride 1024
    for (int kt = 0; kt < 4; ++kt) {
      v16bf a = load_frag_rm(qbase, 1024, kt * 32, lane);
      v16bf b[8];
      #pragma unroll
      for (int t = 0; t < 8; ++t) {
        int st = wv * 8 + t;
        const unsigned short* kb =
            k_bf + (((size_t)bb * 8 + hh) * 1024 + st * 16) * 128;
        b[t] = load_frag_rm(kb, 128, kt * 32, lane);  // K contiguous in d
      }
      #pragma unroll
      for (int t = 0; t < 8; ++t)
        acc[t] = wmma_bf16(a, b[t], acc[t]);
    }
    #pragma unroll
    for (int t = 0; t < 8; ++t) {
      int s = (wv * 8 + t) * 16 + lanel;
      #pragma unroll
      for (int r = 0; r < 8; ++r)
        sc[(r + 8 * laneh) * 1032 + s] = acc[t][r];
    }
  }
  __syncthreads();

  // ---- softmax over s (rows = g)
  {
    int row = threadIdx.x >> 4, c0 = threadIdx.x & 15;
    float m = -1e30f;
    for (int c = c0; c < 1024; c += 16) m = fmaxf(m, sc[row * 1032 + c]);
    red[row * 16 + c0] = m;
    __syncthreads();
    if (threadIdx.x < 16) {
      float mm = -1e30f;
      for (int k = 0; k < 16; ++k) mm = fmaxf(mm, red[threadIdx.x * 16 + k]);
      rowstat[threadIdx.x] = mm;
    }
    __syncthreads();
    float rmax = rowstat[row];
    float ssum = 0.f;
    for (int c = c0; c < 1024; c += 16) {
      float e = __expf(sc[row * 1032 + c] - rmax);
      sc[row * 1032 + c] = e;
      ssum += e;
    }
    red[row * 16 + c0] = ssum;
    __syncthreads();
    if (threadIdx.x < 16) {
      float s = 0.f;
      for (int k = 0; k < 16; ++k) s += red[threadIdx.x * 16 + k];
      rowstat[16 + threadIdx.x] = s;
    }
    __syncthreads();
    float rinv = 1.f / rowstat[16 + row];
    for (int c = c0; c < 1024; c += 16)
      attn[row * 1040 + c] = f2bf(sc[row * 1032 + c] * rinv);
  }
  __syncthreads();

  // ---- o = attn[16,1024] @ V[1024,128]; wave w owns d-tile w.
  // 4 independent accumulators break the 32-deep WMMA->WMMA RAW chain.
  {
    v8f pacc[4];
    #pragma unroll
    for (int p = 0; p < 4; ++p)
      for (int z = 0; z < 8; ++z) pacc[p][z] = 0.f;
    const unsigned short* vb =
        v_t + (((size_t)bb * 8 + hh) * 128 + wv * 16) * 1024;
    for (int ks = 0; ks < 32; ks += 4) {
      v16bf a[4], b[4];
      #pragma unroll
      for (int p = 0; p < 4; ++p) {
        a[p] = load_frag_rm(attn, 1040, (ks + p) * 32, lane);
        b[p] = load_frag_rm(vb, 1024, (ks + p) * 32, lane);
      }
      #pragma unroll
      for (int p = 0; p < 4; ++p)
        pacc[p] = wmma_bf16(a[p], b[p], pacc[p]);
    }
    v8f acc;
    #pragma unroll
    for (int z = 0; z < 8; ++z)
      acc[z] = (pacc[0][z] + pacc[1][z]) + (pacc[2][z] + pacc[3][z]);
    #pragma unroll
    for (int r = 0; r < 8; ++r) {
      int g = r + 8 * laneh;
      int n = g * 8 + hh;               // head_wise index n = g*H + h
      int d = wv * 16 + lanel;
      hw_bf[((size_t)bb * 128 + n) * 128 + d] = f2bf(acc[r]);
    }
  }
}

// ---------------------------------------------------------------------------
// Apre/Bpre = hw[1024,128] @ packed iW0-half[128,256] (+optional ib0), f32 out
// Block = 16 rows; 8 waves x 2 col-tiles.
// ---------------------------------------------------------------------------
__global__ void k_pregemm(const unsigned short* __restrict__ hw_bf,
                          const unsigned short* __restrict__ Wpack,
                          const float* __restrict__ bias,
                          float* __restrict__ outp) {
  const int lane = threadIdx.x & 31, wv = threadIdx.x >> 5;
  const int laneh = lane >> 4, lanel = lane & 15;
  const int m0 = blockIdx.x * 16;
  v8f acc[2];
  #pragma unroll
  for (int t = 0; t < 2; ++t)
    for (int z = 0; z < 8; ++z) acc[t][z] = 0.f;
  const unsigned short* abase = hw_bf + (size_t)m0 * 128;
  for (int kt = 0; kt < 4; ++kt) {
    v16bf a = load_frag_rm(abase, 128, kt * 32, lane);
    v16bf b[2];
    #pragma unroll
    for (int t = 0; t < 2; ++t)
      b[t] = load_frag_packed(Wpack, 4, wv * 2 + t, kt, lane);
    #pragma unroll
    for (int t = 0; t < 2; ++t)
      acc[t] = wmma_bf16(a, b[t], acc[t]);
  }
  #pragma unroll
  for (int t = 0; t < 2; ++t) {
    int n = (wv * 2 + t) * 16 + lanel;
    float bv = bias ? bias[n] : 0.f;
    #pragma unroll
    for (int r = 0; r < 8; ++r)
      outp[(size_t)(m0 + r + 8 * laneh) * 256 + n] = acc[t][r] + bv;
  }
}

// ---------------------------------------------------------------------------
// RelationPool core. Block = one (b, 8x8 pair tile) -> 64 pair rows.
// r0 = relu(Apre[j]+Bpre[i]) (bf16 in LDS) -> GEMM iW1 + relu -> GEMM iW2 +
// relu -> column sums accumulated into agg[b,256].
// ---------------------------------------------------------------------------
__global__ void k_relation(const float* __restrict__ Apre,
                           const float* __restrict__ Bpre,
                           const unsigned short* __restrict__ W1p,
                           const float* __restrict__ b1,
                           const unsigned short* __restrict__ W2p,
                           const float* __restrict__ b2,
                           float* __restrict__ agg) {
  extern __shared__ char smem[];
  unsigned short* buf0 = (unsigned short*)smem;       // 64 x 264 bf16
  unsigned short* buf1 = buf0 + 64 * 264;             // 64 x 264 bf16
  float* colsum = (float*)(buf1 + 64 * 264);          // 256 f32

  const int blk = blockIdx.x;
  const int bb  = blk >> 8;
  const int rem = blk & 255;
  const int i0 = (rem >> 4) * 8, j0 = (rem & 15) * 8;
  const int lane = threadIdx.x & 31, wv = threadIdx.x >> 5;
  const int laneh = lane >> 4, lanel = lane & 15;

  colsum[threadIdx.x] = 0.f;
  // r0 = relu(Apre[b,j] + Bpre[b,i]) for 64 (i,j) pairs
  for (int idx = threadIdx.x; idx < 64 * 256; idx += 256) {
    int p = idx >> 8, c = idx & 255;
    int i = i0 + (p >> 3), j = j0 + (p & 7);
    float v = Apre[((size_t)bb * 128 + j) * 256 + c] +
              Bpre[((size_t)bb * 128 + i) * 256 + c];
    buf0[p * 264 + c] = f2bf(fmaxf(v, 0.f));
  }
  __syncthreads();

  const int mt = wv & 3, ch = wv >> 2;  // 4 row-tiles x 2 col-halves

  // layer 1: buf1 = relu(buf0 @ iW1 + ib1)
  {
    v8f acc[8];
    #pragma unroll
    for (int t = 0; t < 8; ++t)
      for (int z = 0; z < 8; ++z) acc[t][z] = 0.f;
    const unsigned short* abase = buf0 + mt * 16 * 264;
    for (int kt = 0; kt < 8; ++kt) {
      v16bf a = load_frag_rm(abase, 264, kt * 32, lane);
      v16bf b[8];
      #pragma unroll
      for (int t = 0; t < 8; ++t)
        b[t] = load_frag_packed(W1p, 8, ch * 8 + t, kt, lane);
      #pragma unroll
      for (int t = 0; t < 8; ++t)
        acc[t] = wmma_bf16(a, b[t], acc[t]);
    }
    #pragma unroll
    for (int t = 0; t < 8; ++t) {
      int n = (ch * 8 + t) * 16 + lanel;
      float bv = b1[n];
      #pragma unroll
      for (int r = 0; r < 8; ++r)
        buf1[(mt * 16 + r + 8 * laneh) * 264 + n] =
            f2bf(fmaxf(acc[t][r] + bv, 0.f));
    }
  }
  __syncthreads();

  // layer 2: colsum += rowsum(relu(buf1 @ iW2 + ib2))
  {
    v8f acc[8];
    #pragma unroll
    for (int t = 0; t < 8; ++t)
      for (int z = 0; z < 8; ++z) acc[t][z] = 0.f;
    const unsigned short* abase = buf1 + mt * 16 * 264;
    for (int kt = 0; kt < 8; ++kt) {
      v16bf a = load_frag_rm(abase, 264, kt * 32, lane);
      v16bf b[8];
      #pragma unroll
      for (int t = 0; t < 8; ++t)
        b[t] = load_frag_packed(W2p, 8, ch * 8 + t, kt, lane);
      #pragma unroll
      for (int t = 0; t < 8; ++t)
        acc[t] = wmma_bf16(a, b[t], acc[t]);
    }
    #pragma unroll
    for (int t = 0; t < 8; ++t) {
      int n = (ch * 8 + t) * 16 + lanel;
      float bv = b2[n];
      float s8 = 0.f;
      #pragma unroll
      for (int r = 0; r < 8; ++r) s8 += fmaxf(acc[t][r] + bv, 0.f);
      atomicAdd(&colsum[n], s8);   // ds_add_f32
    }
  }
  __syncthreads();
  atomicAdd(&agg[(size_t)bb * 256 + threadIdx.x], colsum[threadIdx.x]);
}

// ---------------------------------------------------------------------------
// Final tiny MLP: agg@iW3 + N^2*ib3 -> outer MLP -> out[8,128]. Single block.
// ---------------------------------------------------------------------------
__global__ void k_final(const float* __restrict__ agg,
                        const float* __restrict__ iW3, const float* __restrict__ ib3,
                        const float* __restrict__ oW0, const float* __restrict__ ob0,
                        const float* __restrict__ oW1, const float* __restrict__ ob1,
                        const float* __restrict__ oW2, const float* __restrict__ ob2,
                        float* __restrict__ outp) {
  __shared__ float sa[8 * 256];
  __shared__ float sb[8 * 256];
  int t = threadIdx.x;
  for (int idx = t; idx < 2048; idx += 256) sa[idx] = agg[idx];
  __syncthreads();
  for (int idx = t; idx < 2048; idx += 256) {   // sb = sa@iW3 + N^2*ib3
    int b = idx >> 8, c = idx & 255;
    float acc = 16384.f * ib3[c];
    for (int k = 0; k < 256; ++k) acc += sa[b * 256 + k] * iW3[k * 256 + c];
    sb[idx] = acc;
  }
  __syncthreads();
  for (int idx = t; idx < 2048; idx += 256) {   // sa = relu(sb@oW0 + ob0)
    int b = idx >> 8, c = idx & 255;
    float acc = ob0[c];
    for (int k = 0; k < 256; ++k) acc += sb[b * 256 + k] * oW0[k * 256 + c];
    sa[idx] = fmaxf(acc, 0.f);
  }
  __syncthreads();
  for (int idx = t; idx < 2048; idx += 256) {   // sb = relu(sa@oW1 + ob1)
    int b = idx >> 8, c = idx & 255;
    float acc = ob1[c];
    for (int k = 0; k < 256; ++k) acc += sa[b * 256 + k] * oW1[k * 256 + c];
    sb[idx] = fmaxf(acc, 0.f);
  }
  __syncthreads();
  for (int idx = t; idx < 1024; idx += 256) {   // out = sb@oW2 + ob2
    int b = idx >> 7, c = idx & 127;
    float acc = ob2[c];
    for (int k = 0; k < 256; ++k) acc += sb[b * 256 + k] * oW2[k * 128 + c];
    outp[idx] = acc;
  }
}

// ---------------------------------------------------------------------------
// Launch
// ---------------------------------------------------------------------------
extern "C" void kernel_launch(void* const* d_in, const int* in_sizes, int n_in,
                              void* d_out, int out_size, void* d_ws, size_t ws_size,
                              hipStream_t stream) {
  const float* x      = (const float*)d_in[0];
  const float* guides = (const float*)d_in[1];
  const float* Wq = (const float*)d_in[2];  const float* bq = (const float*)d_in[3];
  const float* Wk = (const float*)d_in[4];  const float* bk = (const float*)d_in[5];
  const float* Wv = (const float*)d_in[6];  const float* bv = (const float*)d_in[7];
  const float* iW0 = (const float*)d_in[8];  const float* ib0 = (const float*)d_in[9];
  const float* iW1 = (const float*)d_in[10]; const float* ib1 = (const float*)d_in[11];
  const float* iW2 = (const float*)d_in[12]; const float* ib2 = (const float*)d_in[13];
  const float* iW3 = (const float*)d_in[14]; const float* ib3 = (const float*)d_in[15];
  const float* oW0 = (const float*)d_in[16]; const float* ob0 = (const float*)d_in[17];
  const float* oW1 = (const float*)d_in[18]; const float* ob1 = (const float*)d_in[19];
  const float* oW2 = (const float*)d_in[20]; const float* ob2 = (const float*)d_in[21];
  (void)in_sizes; (void)n_in; (void)out_size; (void)ws_size;

  char* ws = (char*)d_ws;
  size_t off = 0;
  auto alloc = [&](size_t bytes) -> char* {
    char* p = ws + off;
    off += (bytes + 255) & ~(size_t)255;
    return p;
  };
  unsigned short* x_bf  = (unsigned short*)alloc((size_t)8192 * 128 * 2);
  unsigned short* Wk_p  = (unsigned short*)alloc((size_t)128 * 1024 * 2);
  unsigned short* Wv_p  = (unsigned short*)alloc((size_t)128 * 1024 * 2);
  unsigned short* W0a_p = (unsigned short*)alloc((size_t)128 * 256 * 2);
  unsigned short* W0b_p = (unsigned short*)alloc((size_t)128 * 256 * 2);
  unsigned short* W1_p  = (unsigned short*)alloc((size_t)256 * 256 * 2);
  unsigned short* W2_p  = (unsigned short*)alloc((size_t)256 * 256 * 2);
  unsigned short* q_bf  = (unsigned short*)alloc((size_t)16 * 1024 * 2);
  unsigned short* k_bf  = (unsigned short*)alloc((size_t)8 * 8 * 1024 * 128 * 2);
  unsigned short* v_t   = (unsigned short*)alloc((size_t)8 * 8 * 1024 * 128 * 2);
  unsigned short* hw_bf = (unsigned short*)alloc((size_t)8 * 128 * 128 * 2);
  float* Apre = (float*)alloc((size_t)8 * 128 * 256 * 4);
  float* Bpre = (float*)alloc((size_t)8 * 128 * 256 * 4);
  float* agg  = (float*)alloc((size_t)8 * 256 * 4);

  k_cvt_bf16<<<4096, 256, 0, stream>>>(x, x_bf, 8192 * 128);
  k_pack_w<<<512, 256, 0, stream>>>(Wk, 1024, 128, 1024, Wk_p);
  k_pack_w<<<512, 256, 0, stream>>>(Wv, 1024, 128, 1024, Wv_p);
  k_pack_w<<<128, 256, 0, stream>>>(iW0, 256, 128, 256, W0a_p);
  k_pack_w<<<128, 256, 0, stream>>>(iW0 + 128 * 256, 256, 128, 256, W0b_p);
  k_pack_w<<<256, 256, 0, stream>>>(iW1, 256, 256, 256, W1_p);
  k_pack_w<<<256, 256, 0, stream>>>(iW2, 256, 256, 256, W2_p);
  k_qproj<<<64, 256, 0, stream>>>(guides, Wq, bq, q_bf);

  k_proj<<<512, 256, 0, stream>>>(x_bf, Wk_p, bk, k_bf, 0);
  k_proj<<<512, 256, 0, stream>>>(x_bf, Wv_p, bv, v_t, 1);

  k_attn<<<64, 256, 100480, stream>>>(q_bf, k_bf, v_t, hw_bf);

  k_pregemm<<<64, 256, 0, stream>>>(hw_bf, W0a_p, ib0, Apre);
  k_pregemm<<<64, 256, 0, stream>>>(hw_bf, W0b_p, nullptr, Bpre);

  k_zero<<<8, 256, 0, stream>>>(agg, 2048);
  k_relation<<<2048, 256, 68608, stream>>>(Apre, Bpre, W1_p, ib1, W2_p, ib2, agg);

  k_final<<<1, 256, 0, stream>>>(agg, iW3, ib3, oW0, ob0, oW1, ob1, oW2, ob2,
                                 (float*)d_out);
}
